// MultiHeadAttention_26972394619318
// MI455X (gfx1250) — matrix operand order repaired and checked by reading, not compiled, not run on hardware
//
#include <hip/hip_runtime.h>

// Problem constants (match reference)
#define B_  2
#define N_  2048
#define D_  512
#define H_  8
#define HS_ 64

typedef __attribute__((ext_vector_type(16))) _Float16 v16h;
typedef __attribute__((ext_vector_type(8)))  _Float16 v8h;
typedef __attribute__((ext_vector_type(4)))  _Float16 v4h;
typedef __attribute__((ext_vector_type(8)))  float    v8f;
typedef __attribute__((ext_vector_type(4)))  float    f32x4;
typedef unsigned int u32x4 __attribute__((ext_vector_type(4)));
typedef int          i32x8 __attribute__((ext_vector_type(8)));
typedef int          i32x4 __attribute__((ext_vector_type(4)));

#if __has_builtin(__builtin_amdgcn_tensor_load_to_lds)
#define USE_TDM 1
#else
#define USE_TDM 0
#endif

// ---------------------------------------------------------------------------
// WMMA fragment loaders (CDNA5 16-bit layouts, ISA 7.12.2)
// A (16x32): lane L -> row M=L%16; K halves {0..7,16..23} (L<16) /
//            {8..15,24..31} (L>=16): two 16B runs 32B apart.
// B (32x16): lane L -> col N=L%16; K halves {0..7,16..23} / {8..15,24..31}.
//            B tiles are staged column-major ([n][k]) in LDS.
// ---------------------------------------------------------------------------
__device__ __forceinline__ v16h load_frag_a(const _Float16* lds, int stride,
                                            int row, int colbase, int lane) {
  const int hi = (lane >> 4) & 1;
  const _Float16* p = lds + row * stride + colbase + hi * 8;
  v8h x0 = *(const v8h*)(p);
  v8h x1 = *(const v8h*)(p + 16);
  v16h r;
#pragma unroll
  for (int i = 0; i < 8; ++i) { r[i] = x0[i]; r[i + 8] = x1[i]; }
  return r;
}

__device__ __forceinline__ v16h load_frag_b(const _Float16* lds, int stride,
                                            int col, int colbase, int lane) {
  const int hi = (lane >> 4) & 1;
  const _Float16* p = lds + col * stride + colbase + hi * 8;
  v8h x0 = *(const v8h*)(p);
  v8h x1 = *(const v8h*)(p + 16);
  v16h r;
#pragma unroll
  for (int i = 0; i < 8; ++i) { r[i] = x0[i]; r[i + 8] = x1[i]; }
  return r;
}

__device__ __forceinline__ v8f wmma_f16(v16h a, v16h b, v8f c) {
  return __builtin_amdgcn_wmma_f32_16x16x32_f16(false, a, false, b,
                                                (short)0, c, false, false);
}

__device__ __forceinline__ unsigned pack2(_Float16 lo, _Float16 hi) {
  union { _Float16 h[2]; unsigned u; } p;
  p.h[0] = lo; p.h[1] = hi;
  return p.u;
}

// ---------------------------------------------------------------------------
// Flash attention: 64 query rows / block, key blocks of 64, online softmax
// (4 lanes per row). Each wave owns TWO 16x16 tiles (row tiles rt0 and rt0+2,
// same column tile), sharing each B fragment across both WMMAs.
// K block is DMA'd by the Tensor Data Mover (D# pad fields produce the
// 72-half LDS pitch); the tensorcnt wait is deferred until after wave 0's
// V-transpose work so the DMA overlaps ALU staging.
// ---------------------------------------------------------------------------
__global__ void __launch_bounds__(256)
attn_kernel(const _Float16* __restrict__ Q, const _Float16* __restrict__ K,
            const _Float16* __restrict__ V, _Float16* __restrict__ MH) {
  __shared__ __align__(16) _Float16 Qs[64][72];   // [qrow][hs]
  __shared__ __align__(16) _Float16 Ks[64][72];   // [key][hs]
  __shared__ __align__(16) _Float16 Vt[64][72];   // [hs][key]
  __shared__ __align__(16) _Float16 Ps[64][72];   // probs f16
  __shared__ float Sb[64][65];                    // logits f32
  __shared__ float red[256];
  __shared__ float mrow[64], lrow[64], crow[64];

  const int tid  = threadIdx.x;
  const int lane = tid & 31;
  const int wave = tid >> 5;
  const int qblk = blockIdx.x % (N_ / 64);
  const int bh   = blockIdx.x / (N_ / 64);
  const int b = bh / H_, h = bh % H_;

  const _Float16* Qp = Q + ((size_t)bh * N_ + (size_t)qblk * 64) * HS_;
  const _Float16* Kp = K + (size_t)bh * N_ * HS_;
  const _Float16* Vp = V + (size_t)bh * N_ * HS_;

#pragma unroll
  for (int i = 0; i < 2; ++i) {                   // stage Q once (v8h copies)
    int e = tid + i * 256;
    int r = e >> 3, cg = (e & 7) * 8;
    *(v8h*)&Qs[r][cg] = *(const v8h*)(Qp + (size_t)r * HS_ + cg);
  }
  if (tid < 64) { mrow[tid] = -1e30f; lrow[tid] = 0.f; }

#if USE_TDM
  // D# group 1: mask=0, data_size=2B(code1), pad_enable, pad_interval=32DW
  // (code 4), pad_amount=4DW (code 3) -> LDS rows of 64 halves + 8 pad = 72.
  i32x8 g1;
  g1[0] = (1 << 16) | (1 << 20) | (4 << 22) | (3 << 25);
  g1[1] = (HS_ & 0xffff) << 16;                   // tensor_dim0 = 64
  g1[2] = (int)((unsigned)(N_ & 0xffff) << 16);   // tensor_dim1 = 2048 (lo16)
  g1[3] = (64 << 16);                             // dim1 hi=0, tile_dim0 = 64
  g1[4] = 64;                                     // tile_dim1 = 64
  g1[5] = HS_;                                    // tensor_dim0_stride = 64
  g1[6] = 0; g1[7] = 0;
  i32x4 g2 = {0, 0, 0, 0}, g3 = {0, 0, 0, 0};
  const unsigned ldsKs = (unsigned)(size_t)(&Ks[0][0]);
#endif

  const int rt0 = wave >> 2;        // row tiles rt0 and rt0+2
  const int ct  = wave & 3;         // key tile (S) / hs tile (O)
  v8f oacc0 = {}, oacc1 = {};

  const int vrp = (tid >> 3) * 2;   // V staging: row pair
  const int vcg = (tid & 7) * 8;    // col group of 8

  for (int kb = 0; kb < N_ / 64; ++kb) {
    const _Float16* Kblk = Kp + (size_t)kb * 64 * HS_;
    const _Float16* Vblk = Vp + (size_t)kb * 64 * HS_;

#if USE_TDM
    if (wave == 0) {                // issue TDM DMA of K tile (64x64 f16)
      unsigned long long ga = (unsigned long long)(size_t)Kblk;
      u32x4 g0;
      g0[0] = 1u;                                  // count=1 valid descriptor
      g0[1] = ldsKs;                               // lds_addr
      g0[2] = (unsigned)(ga & 0xffffffffu);        // global_addr[31:0]
      g0[3] = (unsigned)((ga >> 32) & 0x1ffffffu)  // global_addr[56:32]
              | 0x80000000u;                       // type=2 ("image")
#if defined(__clang_major__) && (__clang_major__ >= 23)
      i32x8 gpad = {0, 0, 0, 0, 0, 0, 0, 0};
      __builtin_amdgcn_tensor_load_to_lds(g0, g1, g2, g3, gpad, 0);
#else
      __builtin_amdgcn_tensor_load_to_lds(g0, g1, g2, g3, 0);
#endif
    }
#else
#pragma unroll
    for (int i = 0; i < 2; ++i) {   // fallback: manual K copy (v8h)
      int e = tid + i * 256;
      int r = e >> 3, cg = (e & 7) * 8;
      *(v8h*)&Ks[r][cg] = *(const v8h*)(Kblk + (size_t)r * HS_ + cg);
    }
#endif
    {                               // V transpose: 2 x b128 load, 8 x b32 store
      v8h a0 = *(const v8h*)(Vblk + (size_t)vrp * HS_ + vcg);
      v8h a1 = *(const v8h*)(Vblk + (size_t)(vrp + 1) * HS_ + vcg);
#pragma unroll
      for (int j = 0; j < 8; ++j)
        *(unsigned*)&Vt[vcg + j][vrp] = pack2(a0[j], a1[j]);
    }
    if (kb + 1 < N_ / 64) {         // WGP-scope prefetch of next blocks
      __builtin_prefetch(Vblk + 64 * HS_ + (size_t)vrp * HS_ + vcg, 0, 3);
      __builtin_prefetch(Kblk + 64 * HS_ + (size_t)vrp * HS_ + vcg, 0, 3);
    }
#if USE_TDM
    if (wave == 0)                  // drain DMA only after overlap work
      __builtin_amdgcn_s_wait_tensorcnt(0);
#endif
    __syncthreads();

    // ---- S = Q * K^T : two row tiles per wave, B fragment shared ----
    v8f sacc0 = {}, sacc1 = {};
#pragma unroll
    for (int c0 = 0; c0 < 64; c0 += 32) {
      v16h bf = load_frag_b(&Ks[0][0], 72, ct * 16 + (lane & 15), c0, lane);
      v16h a0 = load_frag_a(&Qs[0][0], 72, rt0 * 16 + (lane & 15), c0, lane);
      sacc0 = wmma_f16(a0, bf, sacc0);
      v16h a1 = load_frag_a(&Qs[0][0], 72, (rt0 + 2) * 16 + (lane & 15), c0, lane);
      sacc1 = wmma_f16(a1, bf, sacc1);
    }
#pragma unroll
    for (int r = 0; r < 8; ++r) {
      int ri = r + ((lane >> 4) << 3);
      Sb[rt0 * 16 + ri][ct * 16 + (lane & 15)]       = sacc0[r];
      Sb[(rt0 + 2) * 16 + ri][ct * 16 + (lane & 15)] = sacc1[r];
    }
    __syncthreads();

    // ---- online softmax: 4 lanes per query row, 16 elems each ----
    {
      int r = tid >> 2, c0 = (tid & 3) * 16;
      float m = Sb[r][c0];
#pragma unroll
      for (int j = 1; j < 16; ++j) m = fmaxf(m, Sb[r][c0 + j]);
      red[tid] = m;
    }
    __syncthreads();
    if (tid < 64) {
      float m_old = mrow[tid];
      float mb = m_old;
#pragma unroll
      for (int j = 0; j < 4; ++j) mb = fmaxf(mb, red[tid * 4 + j]);
      mrow[tid] = mb;
      crow[tid] = __expf(m_old - mb);
    }
    __syncthreads();
    {
      int r = tid >> 2, c0 = (tid & 3) * 16;
      float mb = mrow[r];
      float s = 0.f;
#pragma unroll
      for (int j = 0; j < 16; ++j) {
        float p = __expf(Sb[r][c0 + j] - mb);
        Ps[r][c0 + j] = (_Float16)p;
        s += p;
      }
      red[tid] = s;
    }
    __syncthreads();
    if (tid < 64) {
      float s = 0.f;
#pragma unroll
      for (int j = 0; j < 4; ++j) s += red[tid * 4 + j];
      lrow[tid] = lrow[tid] * crow[tid] + s;
    }

    // ---- O = O*corr + P * V : two row tiles per wave ----
#pragma unroll
    for (int r = 0; r < 8; ++r) {
      int ri = r + ((lane >> 4) << 3);
      oacc0[r] *= crow[rt0 * 16 + ri];
      oacc1[r] *= crow[(rt0 + 2) * 16 + ri];
    }
#pragma unroll
    for (int c0 = 0; c0 < 64; c0 += 32) {
      v16h bf = load_frag_b(&Vt[0][0], 72, ct * 16 + (lane & 15), c0, lane);
      v16h a0 = load_frag_a(&Ps[0][0], 72, rt0 * 16 + (lane & 15), c0, lane);
      oacc0 = wmma_f16(a0, bf, oacc0);
      v16h a1 = load_frag_a(&Ps[0][0], 72, (rt0 + 2) * 16 + (lane & 15), c0, lane);
      oacc1 = wmma_f16(a1, bf, oacc1);
    }
    __syncthreads();
  }

#pragma unroll
  for (int r = 0; r < 8; ++r) {
    int ri  = r + ((lane >> 4) << 3);
    int col = ct * 16 + (lane & 15);
    {
      int rr = rt0 * 16 + ri;
      int qrow = qblk * 64 + rr;
      MH[((size_t)b * N_ + qrow) * (H_ * HS_) + h * HS_ + col] =
          (_Float16)(oacc0[r] / lrow[rr]);
    }
    {
      int rr = (rt0 + 2) * 16 + ri;
      int qrow = qblk * 64 + rr;
      MH[((size_t)b * N_ + qrow) * (H_ * HS_) + h * HS_ + col] =
          (_Float16)(oacc1[r] / lrow[rr]);
    }
  }
}

// ---------------------------------------------------------------------------
// Per-head input projection [N x D] x [D x HS] -> f16 [B,H,N,HS].
// Block = 64 rows x 64 cols; wave = 16x32 strip (2 accumulators).
// ---------------------------------------------------------------------------
__global__ void __launch_bounds__(256)
proj_kernel(const float* __restrict__ X, const float* __restrict__ W,
            _Float16* __restrict__ Out, float scale) {
  __shared__ __align__(16) _Float16 Xs[64][40];   // [m][k]
  __shared__ __align__(16) _Float16 Ws[64][40];   // [n][k] transposed

  const int tid  = threadIdx.x;
  const int lane = tid & 31;
  const int wave = tid >> 5;
  const int rowblk = blockIdx.x % (N_ / 64);
  const int bh     = blockIdx.x / (N_ / 64);

  const float* Xb = X + (size_t)(bh / H_) * N_ * D_ + (size_t)rowblk * 64 * D_;
  const float* Wh = W + (size_t)(bh % H_) * D_ * HS_;
  _Float16*    Ob = Out + (size_t)bh * N_ * HS_ + (size_t)rowblk * 64 * HS_;

  const int rt  = wave >> 1;
  const int cth = (wave & 1) * 2;
  v8f acc0 = {}, acc1 = {};

  const int xr  = tid >> 3;
  const int xc4 = (tid & 7) * 4;
  const int wk  = (tid >> 4) * 2;
  const int wng = (tid & 15) * 4;

  for (int k0 = 0; k0 < D_; k0 += 32) {
#pragma unroll
    for (int i = 0; i < 2; ++i) {
      int r = xr + i * 32;
      f32x4 x = *(const f32x4*)(Xb + (size_t)r * D_ + k0 + xc4);
      v4h hh;
      hh[0] = (_Float16)(x.x * scale); hh[1] = (_Float16)(x.y * scale);
      hh[2] = (_Float16)(x.z * scale); hh[3] = (_Float16)(x.w * scale);
      *(v4h*)&Xs[r][xc4] = hh;
    }
    {
      f32x4 w0 = *(const f32x4*)(Wh + (size_t)(k0 + wk) * HS_ + wng);
      f32x4 w1 = *(const f32x4*)(Wh + (size_t)(k0 + wk + 1) * HS_ + wng);
#pragma unroll
      for (int j = 0; j < 4; ++j)
        *(unsigned*)&Ws[wng + j][wk] = pack2((_Float16)w0[j], (_Float16)w1[j]);
    }
    if (k0 + 32 < D_) {
      __builtin_prefetch(Xb + (size_t)xr * D_ + k0 + 32 + xc4, 0, 3);
      __builtin_prefetch(Wh + (size_t)(k0 + 32 + wk) * HS_ + wng, 0, 3);
    }
    __syncthreads();
    v16h a  = load_frag_a(&Xs[0][0], 40, rt * 16 + (lane & 15), 0, lane);
    v16h b0 = load_frag_b(&Ws[0][0], 40, cth * 16 + (lane & 15), 0, lane);
    acc0 = wmma_f16(a, b0, acc0);
    v16h b1 = load_frag_b(&Ws[0][0], 40, (cth + 1) * 16 + (lane & 15), 0, lane);
    acc1 = wmma_f16(a, b1, acc1);
    __syncthreads();
  }
#pragma unroll
  for (int r = 0; r < 8; ++r) {
    int row = rt * 16 + r + ((lane >> 4) << 3);
    Ob[(size_t)row * HS_ + cth * 16 + (lane & 15)]       = (_Float16)acc0[r];
    Ob[(size_t)row * HS_ + (cth + 1) * 16 + (lane & 15)] = (_Float16)acc1[r];
  }
}

// ---------------------------------------------------------------------------
// Output projection [B*N, 512] x [512, 512] -> f32 [B,N,D].
// ---------------------------------------------------------------------------
__global__ void __launch_bounds__(256)
outproj_kernel(const _Float16* __restrict__ MH, const float* __restrict__ Wp,
               float* __restrict__ Out) {
  __shared__ __align__(16) _Float16 As[64][40];
  __shared__ __align__(16) _Float16 Bs[64][40];

  const int tid  = threadIdx.x;
  const int lane = tid & 31;
  const int wave = tid >> 5;
  const int nrb  = (B_ * N_) / 64;
  const int rowblk = blockIdx.x % nrb;
  const int cb     = blockIdx.x / nrb;

  const _Float16* Ap = MH + (size_t)rowblk * 64 * D_;
  const int rt  = wave >> 1;
  const int cth = (wave & 1) * 2;
  v8f acc0 = {}, acc1 = {};

  const int ar  = tid >> 2, acg = (tid & 3) * 8;
  const int wk  = (tid >> 4) * 2;
  const int wng = (tid & 15) * 4;

  for (int k0 = 0; k0 < D_; k0 += 32) {
    *(v8h*)&As[ar][acg] = *(const v8h*)(Ap + (size_t)ar * D_ + k0 + acg);
    {
      f32x4 w0 = *(const f32x4*)(Wp + (size_t)(k0 + wk) * D_ + cb * 64 + wng);
      f32x4 w1 = *(const f32x4*)(Wp + (size_t)(k0 + wk + 1) * D_ + cb * 64 + wng);
#pragma unroll
      for (int j = 0; j < 4; ++j)
        *(unsigned*)&Bs[wng + j][wk] = pack2((_Float16)w0[j], (_Float16)w1[j]);
    }
    if (k0 + 32 < D_)
      __builtin_prefetch(Wp + (size_t)(k0 + 32 + wk) * D_ + cb * 64 + wng, 0, 3);
    __syncthreads();
    v16h a  = load_frag_a(&As[0][0], 40, rt * 16 + (lane & 15), 0, lane);
    v16h b0 = load_frag_b(&Bs[0][0], 40, cth * 16 + (lane & 15), 0, lane);
    acc0 = wmma_f16(a, b0, acc0);
    v16h b1 = load_frag_b(&Bs[0][0], 40, (cth + 1) * 16 + (lane & 15), 0, lane);
    acc1 = wmma_f16(a, b1, acc1);
    __syncthreads();
  }
#pragma unroll
  for (int r = 0; r < 8; ++r) {
    int row = rowblk * 64 + rt * 16 + r + ((lane >> 4) << 3);
    Out[(size_t)row * D_ + cb * 64 + cth * 16 + (lane & 15)]       = acc0[r];
    Out[(size_t)row * D_ + cb * 64 + (cth + 1) * 16 + (lane & 15)] = acc1[r];
  }
}

// ---------------------------------------------------------------------------
extern "C" void kernel_launch(void* const* d_in, const int* in_sizes, int n_in,
                              void* d_out, int out_size, void* d_ws, size_t ws_size,
                              hipStream_t stream) {
  const float* query = (const float*)d_in[0];
  const float* key   = (const float*)d_in[1];
  const float* value = (const float*)d_in[2];
  const float* Wq    = (const float*)d_in[3];
  const float* Wk    = (const float*)d_in[4];
  const float* Wv    = (const float*)d_in[5];
  const float* Wp    = (const float*)d_in[6];
  float* out = (float*)d_out;

  // Workspace (f16): Qh | Kh | Vh [B,H,N,HS], MH [B*N, H*HS]  (16 MB total)
  char* ws = (char*)d_ws;
  const size_t perHead = (size_t)B_ * H_ * N_ * HS_ * sizeof(_Float16);
  _Float16* Qh = (_Float16*)(ws);
  _Float16* Kh = (_Float16*)(ws + perHead);
  _Float16* Vh = (_Float16*)(ws + 2 * perHead);
  _Float16* MH = (_Float16*)(ws + 3 * perHead);

  dim3 blk(256);
  dim3 gproj(B_ * H_ * (N_ / 64));               // 512 blocks
  const float qscale = 0.125f;                   // 1/sqrt(HS)
  proj_kernel<<<gproj, blk, 0, stream>>>(query, Wq, Qh, qscale);
  proj_kernel<<<gproj, blk, 0, stream>>>(key,   Wk, Kh, 1.0f);
  proj_kernel<<<gproj, blk, 0, stream>>>(value, Wv, Vh, 1.0f);

  dim3 gattn(B_ * H_ * (N_ / 64));               // 512 blocks
  attn_kernel<<<gattn, blk, 0, stream>>>(Qh, Kh, Vh, MH);

  dim3 gout(((B_ * N_) / 64) * (D_ / 64));       // 512 blocks
  outproj_kernel<<<gout, blk, 0, stream>>>(MH, Wp, out);
}
